// MemN2NDialog_24275155157603
// MI455X (gfx1250) — matrix operand order repaired and checked by reading, not compiled, not run on hardware
//
#include <hip/hip_runtime.h>

// Problem constants (from reference)
constexpr int NB = 64;     // batch
constexpr int NM = 200;    // memories per batch
constexpr int NS = 50;     // tokens per sentence
constexpr int NC = 10000;  // candidates
constexpr int NE = 64;     // embedding dim
constexpr int ND = 128;    // 2*E (concat word+mask)
constexpr int NHOPS = 3;

typedef _Float16 v16h __attribute__((ext_vector_type(16)));
typedef float    v8f  __attribute__((ext_vector_type(8)));

// ---------------------------------------------------------------------------
// Gather + sum embeddings for one output row per block.
// out[row, 0:64]   = sum_s emb[idx_word[row,s], :]
// out[row, 64:128] = sum_s emb[idx_mask[row,s], :]
// 128 threads: t<64 handles word part dim t, t>=64 handles mask part dim t-64.
// Lanes read 64 consecutive floats of an embedding row -> coalesced 128B/wave.
// ---------------------------------------------------------------------------
__global__ __launch_bounds__(128)
void embed_sum_kernel(const int* __restrict__ idx_word,
                      const int* __restrict__ idx_mask,
                      const float* __restrict__ emb,
                      float* __restrict__ out) {
  const int row = blockIdx.x;
  const int t   = threadIdx.x;
  const int* idx = (t < NE) ? idx_word : idx_mask;
  const int col  = t & (NE - 1);
  const int* ip  = idx + (long)row * NS;
  float acc = 0.0f;
  #pragma unroll 5
  for (int s = 0; s < NS; ++s) {
    const int id = ip[s];                 // wave-uniform broadcast load
    acc += emb[(long)id * NE + col];      // coalesced gather of one emb row
  }
  out[(long)row * ND + t] = acc;
}

// ---------------------------------------------------------------------------
// 3 memory hops for one batch element per block (64 blocks, 128 threads).
// attn = softmax(m[b] @ u); o = attn @ m[b]; u = u @ Hw^T + Hb + o
// ---------------------------------------------------------------------------
__global__ __launch_bounds__(128)
void hops_kernel(const float* __restrict__ m_mem,  // [NB, NM, ND]
                 const float* __restrict__ u_in,   // [NB, ND]
                 const float* __restrict__ Hw,     // [ND, ND]
                 const float* __restrict__ Hb,     // [ND]
                 float* __restrict__ u_out) {      // [NB, ND]
  __shared__ float u_s[ND];
  __shared__ float attn[NM];
  __shared__ float mx_s, inv_sum_s;

  const int b = blockIdx.x;
  const int t = threadIdx.x;
  const float* mb = m_mem + (long)b * NM * ND;

  u_s[t] = u_in[(long)b * ND + t];
  __syncthreads();

  for (int hop = 0; hop < NHOPS; ++hop) {
    // scores[mi] = dot(m[b,mi,:], u)
    for (int mi = t; mi < NM; mi += 128) {
      const float* mr = mb + (long)mi * ND;
      float dot = 0.0f;
      for (int k = 0; k < ND; ++k) dot += mr[k] * u_s[k];
      attn[mi] = dot;
    }
    __syncthreads();
    if (t == 0) {                       // serial max: NM=200, negligible
      float mx = attn[0];
      for (int i = 1; i < NM; ++i) mx = fmaxf(mx, attn[i]);
      mx_s = mx;
    }
    __syncthreads();
    for (int mi = t; mi < NM; mi += 128) attn[mi] = expf(attn[mi] - mx_s);
    __syncthreads();
    if (t == 0) {
      float s = 0.0f;
      for (int i = 0; i < NM; ++i) s += attn[i];
      inv_sum_s = 1.0f / s;
    }
    __syncthreads();

    // o[t] = (1/sum) * sum_mi attn[mi] * m[b,mi,t]   (coalesced across t)
    float o = 0.0f;
    for (int mi = 0; mi < NM; ++mi) o += attn[mi] * mb[(long)mi * ND + t];
    o *= inv_sum_s;

    // linear: u_new[t] = sum_k Hw[t,k]*u[k] + Hb[t] + o
    float lin = Hb[t];
    for (int k = 0; k < ND; ++k) lin += Hw[(long)t * ND + k] * u_s[k];

    __syncthreads();
    u_s[t] = lin + o;
    __syncthreads();
  }

  u_out[(long)b * ND + t] = u_s[t];
}

// ---------------------------------------------------------------------------
// Logits = U [64,128] @ Cand^T [128,10000] via v_wmma_f32_16x16x32_f16.
// 625 blocks x 128 threads; wave w owns output tile rows [16w,16w+16) x
// cols [16*blockIdx.x, +16). K=128 fully unrolled as 4 steps of 32.
//
// f32 accuracy via hi/lo f16 split: x = hi + lo (hi = f16(x), lo exact
// residual); a*b ~= a_lo*b_hi + a_hi*b_lo + a_hi*b_hi, f32 accumulate.
//
// Fragment layouts used here (A and B share contiguous k order k0+16*hgrp+i):
//  A 16x32 f16: lane (hgrp=0) holds K = k0+[0..7] and k0+8+[0..7];
//               lane (hgrp=1) holds K = k0+16+[0..7] and k0+24+[0..7].
//               Element i<8 -> first run, i>=8 -> second run (contiguous!).
//  B 32x16 f16: lane group = column N; element i -> K = k0 + 16*hgrp + i
//               (one contiguous 16-float run; Cand row-major == B col-major).
//  C/D f32 16x16: VGPR r -> M = r + 8*hgrp, N = lane%16.
// ---------------------------------------------------------------------------
__global__ __launch_bounds__(128)
void gemm_wmma_kernel(const float* __restrict__ U,     // [NB, ND]
                      const float* __restrict__ Cand,  // [NC, ND]
                      float* __restrict__ Out) {       // [NB, NC]
  const int wave = threadIdx.x >> 5;
  const int lane = threadIdx.x & 31;
  const int hgrp = lane >> 4;     // lane half-group (0: lanes 0-15, 1: 16-31)
  const int lrow = lane & 15;
  const int rowBase = wave * 16;           // 4 waves cover NB=64 rows
  const int colBase = blockIdx.x * 16;     // 625 blocks cover NC=10000 cols

  v8f acc = {};
  const float* arow = U    + (long)(rowBase + lrow) * ND;   // A row M
  const float* brow = Cand + (long)(colBase + lrow) * ND;   // B column N

  #pragma unroll
  for (int k0 = 0; k0 < ND; k0 += 32) {
    v16h a_hi, a_lo, b_hi, b_lo;

    // A fragment: two contiguous 8-element runs per lane.
    const int abase0 = k0 + (hgrp ? 16 : 0);       // elements 0..7
    const int abase1 = abase0 + 8;                 // elements 8..15
    #pragma unroll
    for (int i = 0; i < 8; ++i) {
      {
        const float x = arow[abase0 + i];
        const _Float16 h = (_Float16)x;
        a_hi[i] = h;
        a_lo[i] = (_Float16)(x - (float)h);
      }
      {
        const float x = arow[abase1 + i];
        const _Float16 h = (_Float16)x;
        a_hi[i + 8] = h;
        a_lo[i + 8] = (_Float16)(x - (float)h);
      }
    }

    // B fragment: one contiguous 16-element run per lane.
    const int bbase = k0 + (hgrp ? 16 : 0);
    #pragma unroll
    for (int i = 0; i < 16; ++i) {
      const float x = brow[bbase + i];
      const _Float16 h = (_Float16)x;
      b_hi[i] = h;
      b_lo[i] = (_Float16)(x - (float)h);
    }

    // Small correction terms first, main term last (best f32 accumulation).
    acc = __builtin_amdgcn_wmma_f32_16x16x32_f16(
        false, a_lo, false, b_hi, (short)0, acc, false, false);
    acc = __builtin_amdgcn_wmma_f32_16x16x32_f16(
        false, a_hi, false, b_lo, (short)0, acc, false, false);
    acc = __builtin_amdgcn_wmma_f32_16x16x32_f16(
        false, a_hi, false, b_hi, (short)0, acc, false, false);
  }

  // f32 C/D 16x16 layout: VGPR r -> M = r + 8*hgrp, N = lane%16.
  #pragma unroll
  for (int r = 0; r < 8; ++r) {
    const int mrow = rowBase + r + (hgrp ? 8 : 0);
    Out[(long)mrow * NC + colBase + lrow] = acc[r];
  }
}

// ---------------------------------------------------------------------------
extern "C" void kernel_launch(void* const* d_in, const int* in_sizes, int n_in,
                              void* d_out, int out_size, void* d_ws, size_t ws_size,
                              hipStream_t stream) {
  const int*   stories         = (const int*)  d_in[0];
  const int*   query           = (const int*)  d_in[1];
  const int*   stories_mask    = (const int*)  d_in[2];
  const int*   query_mask      = (const int*)  d_in[3];
  const int*   candidates      = (const int*)  d_in[4];
  const int*   candidates_mask = (const int*)  d_in[5];
  const float* Aemb            = (const float*)d_in[6];
  const float* Wemb            = (const float*)d_in[7];
  const float* Hw              = (const float*)d_in[8];
  const float* Hb              = (const float*)d_in[9];
  float* out = (float*)d_out;

  // Workspace layout (floats): m_mem | cand | u0 | u1  (~11.7 MB total)
  float* ws    = (float*)d_ws;
  float* m_mem = ws;                                  // [NB*NM, ND]
  float* cand  = m_mem + (size_t)NB * NM * ND;        // [NC, ND]
  float* u0    = cand  + (size_t)NC * ND;             // [NB, ND]
  float* u1    = u0    + (size_t)NB * ND;             // [NB, ND]

  // 1) story memory: 12800 rows
  embed_sum_kernel<<<NB * NM, 128, 0, stream>>>(stories, stories_mask, Aemb, m_mem);
  // 2) query embedding: 64 rows
  embed_sum_kernel<<<NB, 128, 0, stream>>>(query, query_mask, Aemb, u0);
  // 3) candidate embeddings: 10000 rows
  embed_sum_kernel<<<NC, 128, 0, stream>>>(candidates, candidates_mask, Wemb, cand);
  // 4) 3 attention hops
  hops_kernel<<<NB, 128, 0, stream>>>(m_mem, u0, Hw, Hb, u1);
  // 5) logits via WMMA (hi/lo split f16 -> ~f32 accuracy)
  gemm_wmma_kernel<<<NC / 16, 128, 0, stream>>>(u1, cand, out);
}
